// ClauseGNNWithAttention_59880434041168
// MI455X (gfx1250) — matrix operand order repaired and checked by reading, not compiled, not run on hardware
//
#include <hip/hip_runtime.h>

#define HD 128

typedef __bf16 v16bf __attribute__((ext_vector_type(16)));
typedef float  v8f   __attribute__((ext_vector_type(8)));

union AFrag { v16bf v; uint4 q[2]; };   // raw 2x16B -> one 16x32 bf16 A fragment

__device__ __forceinline__ unsigned short f2bf(float f) {
  unsigned int x = __float_as_uint(f);
  x += 0x7FFFu + ((x >> 16) & 1u);   // round-to-nearest-even
  return (unsigned short)(x >> 16);
}

// ---------------- utility fills / converts ----------------
__global__ void fill_f32(float* p, float v, int n) {
  int i = blockIdx.x * blockDim.x + threadIdx.x;
  if (i < n) p[i] = v;
}

// f32 -> bf16, 4 elements per thread
__global__ void to_bf16(const float* __restrict__ in, unsigned short* __restrict__ out, int n4) {
  int i = blockIdx.x * blockDim.x + threadIdx.x;
  if (i >= n4) return;
  float4 v = *(const float4*)(in + (size_t)i * 4);
  ushort4 r;
  r.x = f2bf(v.x); r.y = f2bf(v.y); r.z = f2bf(v.z); r.w = f2bf(v.w);
  *(ushort4*)(out + (size_t)i * 4) = r;
}

// W [K=128][N=128] f32 row-major -> BT [n][k] bf16 (B-fragment lanes read 16 contiguous bf16)
__global__ void prep_w(const float* __restrict__ W, unsigned short* __restrict__ BT) {
  int idx = blockIdx.x * blockDim.x + threadIdx.x;   // idx = n*128 + k
  int n = idx >> 7, k = idx & 127;
  BT[idx] = f2bf(W[k * HD + n]);
}

// ---------------- WMMA GEMM: C[nrows,128] = A[nrows,128](bf16) @ B + (bias) ----------------
// 8 waves/block, each wave owns a 16-column strip and 4 row tiles (64x128 per block).
// K-outer loop: B fragment loaded once per K-step, 4 accumulators held live.
__global__ __launch_bounds__(256)
void gemm_bf16_wmma(const unsigned short* __restrict__ A,     // [nrows][128] bf16 row-major
                    const unsigned short* __restrict__ BT,    // [128][128] bf16, BT[n][k]
                    const float* __restrict__ bias,
                    float* __restrict__ C,
                    int nrows, int has_bias) {
  const int lane  = threadIdx.x & 31;
  const int wave  = threadIdx.x >> 5;   // 0..7 -> 16-column strip
  const int col16 = lane & 15;
  const int half  = lane >> 4;          // 0/1 selects K-half: k = 16*half + s (contiguous order used here)
  const int col   = wave * 16 + col16;
  const int rowBlock = blockIdx.x * 64;

  // Clamped (in-bounds, wave-uniform) row base per tile; garbage tiles never stored.
  int mrow[4];
#pragma unroll
  for (int rt = 0; rt < 4; ++rt) {
    int m0 = rowBlock + rt * 16;
    mrow[rt] = (m0 < nrows) ? m0 : (nrows - 16);
  }

  const unsigned short* bbase = BT + col * HD + half * 16;
  v8f acc[4] = {{}, {}, {}, {}};

#pragma unroll
  for (int ks = 0; ks < 4; ++ks) {
    // B 32x16 bf16 fragment: lane l<16 K 0..15, lane>=16 K 16..31 (16 contiguous bf16)
    v16bf bfrag = *(const v16bf*)(bbase + ks * 32);
#pragma unroll
    for (int rt = 0; rt < 4; ++rt) {
      // A 16x32 bf16 fragment: row = m0+col16, K contiguous [16*half, 16*half+16)
      const unsigned short* ap = A + (size_t)(mrow[rt] + col16) * HD + half * 16 + ks * 32;
      AFrag af;
      af.q[0] = *(const uint4*)(ap);
      af.q[1] = *(const uint4*)(ap + 8);
      acc[rt] = __builtin_amdgcn_wmma_f32_16x16x32_bf16(
          false, af.v, false, bfrag, (short)0, acc[rt], false, false);
    }
  }

  const float bb = has_bias ? bias[col] : 0.0f;
#pragma unroll
  for (int rt = 0; rt < 4; ++rt) {
    const int m0 = rowBlock + rt * 16;
    if (m0 >= nrows) break;                       // nrows is a multiple of 16
    // C/D layout: element r -> row m0 + r + 8*half, col
    float* cp = C + (size_t)(m0 + half * 8) * HD + col;
#pragma unroll
    for (int r = 0; r < 8; ++r)
      cp[(size_t)r * HD] = acc[rt][r] + bb;
  }
}

// ---------------- graph plumbing ----------------
__global__ void degree_count(const int* __restrict__ dst, float* __restrict__ deg, int E) {
  int i = blockIdx.x * blockDim.x + threadIdx.x;
  if (i < E) atomicAdd(&deg[dst[i]], 1.0f);
}

__global__ void make_dinv(const float* __restrict__ deg, float* __restrict__ dinv, int n) {
  int i = blockIdx.x * blockDim.x + threadIdx.x;
  if (i < n) dinv[i] = rsqrtf(deg[i] + 1.0f);   // deg includes self-loop
}

// one wave per edge; lane handles 4 features (float4 gather from L2-resident h)
__global__ __launch_bounds__(256)
void edge_agg(const int* __restrict__ src, const int* __restrict__ dst,
              const float* __restrict__ dinv, const float* __restrict__ h,
              float* __restrict__ agg, int E) {
  int e = blockIdx.x * 8 + (threadIdx.x >> 5);
  if (e >= E) return;
  int lane = threadIdx.x & 31;
  int s = src[e], d = dst[e];
  float nrm = dinv[s] * dinv[d];
  float4 v = *(const float4*)(h + (size_t)s * HD + lane * 4);
  float* ap = agg + (size_t)d * HD + lane * 4;
  atomicAdd(ap + 0, nrm * v.x);
  atomicAdd(ap + 1, nrm * v.y);
  atomicAdd(ap + 2, nrm * v.z);
  atomicAdd(ap + 3, nrm * v.w);
}

// y = agg + hW/deg + b (in-place into agg_y) + per-feature BN partial sums
__global__ __launch_bounds__(128)
void finish_stats(const float* __restrict__ hw, float* __restrict__ agg_y,
                  const float* __restrict__ dinv, const float* __restrict__ bias,
                  float* __restrict__ bnsum, float* __restrict__ bnsumsq,
                  int nrows, int chunk) {
  int f  = threadIdx.x;                // feature
  int i0 = blockIdx.x * chunk;
  int i1 = i0 + chunk; if (i1 > nrows) i1 = nrows;
  float b = bias[f];
  float s = 0.f, ss = 0.f;
  for (int i = i0; i < i1; ++i) {
    float di = dinv[i];
    float y  = agg_y[(size_t)i * HD + f] + hw[(size_t)i * HD + f] * (di * di) + b;
    agg_y[(size_t)i * HD + f] = y;
    s += y; ss += y * y;
  }
  atomicAdd(&bnsum[f], s);
  atomicAdd(&bnsumsq[f], ss);
}

// BN + ReLU; writes f32 activations AND bf16 copy for the next WMMA GEMM
__global__ void bn_apply_relu(const float* __restrict__ y,
                              const float* __restrict__ bnsum,
                              const float* __restrict__ bnsumsq,
                              const float* __restrict__ gamma,
                              const float* __restrict__ beta,
                              float* __restrict__ out,
                              unsigned short* __restrict__ out_bf,
                              int nrows) {
  int idx = blockIdx.x * blockDim.x + threadIdx.x;     // one thread = 4 features of a node
  if (idx >= nrows * 32) return;
  int f0 = (idx & 31) * 4;
  float invN = 1.0f / (float)nrows;
  float4 yv = *(const float4*)(y + (size_t)idx * 4);
  float ov[4] = {yv.x, yv.y, yv.z, yv.w};
  float rs[4];
#pragma unroll
  for (int c = 0; c < 4; ++c) {
    int f = f0 + c;
    float mu   = bnsum[f] * invN;
    float var  = bnsumsq[f] * invN - mu * mu;
    float rstd = rsqrtf(var + 1e-5f);
    rs[c] = fmaxf(gamma[f] * (ov[c] - mu) * rstd + beta[f], 0.0f);
  }
  float4 r; r.x = rs[0]; r.y = rs[1]; r.z = rs[2]; r.w = rs[3];
  *(float4*)(out + (size_t)idx * 4) = r;
  ushort4 rb;
  rb.x = f2bf(rs[0]); rb.y = f2bf(rs[1]); rb.z = f2bf(rs[2]); rb.w = f2bf(rs[3]);
  *(ushort4*)(out_bf + (size_t)idx * 4) = rb;
}

// ---------------- attention head ----------------
__global__ __launch_bounds__(256)
void score_kernel(const float* __restrict__ t, const float* __restrict__ aw2,
                  const float* __restrict__ ab2, float* __restrict__ score, int n) {
  int i = blockIdx.x * 8 + (threadIdx.x >> 5);
  if (i >= n) return;
  int lane = threadIdx.x & 31;
  float4 tv = *(const float4*)(t + (size_t)i * HD + lane * 4);
  float4 wv = *(const float4*)(aw2 + lane * 4);
  float p = tanhf(tv.x) * wv.x + tanhf(tv.y) * wv.y +
            tanhf(tv.z) * wv.z + tanhf(tv.w) * wv.w;
  for (int o = 16; o; o >>= 1) p += __shfl_xor(p, o, 32);
  if (lane == 0) score[i] = p + ab2[0];
}

__device__ __forceinline__ void atomicMaxF(float* addr, float val) {
  int old = __float_as_int(*addr);
  while (val > __int_as_float(old)) {
    int assumed = old;
    old = atomicCAS((int*)addr, assumed, __float_as_int(val));
    if (old == assumed) break;
  }
}

__global__ void seg_max(const float* __restrict__ score, const int* __restrict__ batch,
                        float* __restrict__ m, int n) {
  int i = blockIdx.x * blockDim.x + threadIdx.x;
  if (i < n) atomicMaxF(&m[batch[i]], score[i]);
}

__global__ void seg_expsum(const float* __restrict__ score, const int* __restrict__ batch,
                           const float* __restrict__ m, float* __restrict__ e,
                           float* __restrict__ denom, int n) {
  int i = blockIdx.x * blockDim.x + threadIdx.x;
  if (i < n) {
    int b = batch[i];
    float v = __expf(score[i] - m[b]);
    e[i] = v;
    atomicAdd(&denom[b], v);
  }
}

__global__ __launch_bounds__(256)
void pool_kernel(const float* __restrict__ h, const float* __restrict__ e,
                 const float* __restrict__ denom, const int* __restrict__ batch,
                 float* __restrict__ g, int n) {
  int i = blockIdx.x * 8 + (threadIdx.x >> 5);
  if (i >= n) return;
  int lane = threadIdx.x & 31;
  int b = batch[i];
  float w = e[i] / denom[b];
  float4 hv = *(const float4*)(h + (size_t)i * HD + lane * 4);
  float* gp = g + (size_t)b * HD + lane * 4;
  atomicAdd(gp + 0, w * hv.x);
  atomicAdd(gp + 1, w * hv.y);
  atomicAdd(gp + 2, w * hv.z);
  atomicAdd(gp + 3, w * hv.w);
}

// out[b] = relu(g[b] @ mW1 + mb1) @ mw2 + mb2   (one block per graph)
__global__ __launch_bounds__(128)
void final_mlp(const float* __restrict__ g, const float* __restrict__ mW1,
               const float* __restrict__ mb1, const float* __restrict__ mw2,
               const float* __restrict__ mb2, float* __restrict__ out) {
  __shared__ float gs[HD];
  __shared__ float red[HD];
  int b = blockIdx.x, j = threadIdx.x;
  gs[j] = g[(size_t)b * HD + j];
  __syncthreads();
  float acc = mb1[j];
  for (int k = 0; k < HD; ++k) acc += gs[k] * mW1[k * HD + j];
  red[j] = fmaxf(acc, 0.0f) * mw2[j];
  __syncthreads();
  for (int s = 64; s > 0; s >>= 1) {
    if (j < s) red[j] += red[j + s];
    __syncthreads();
  }
  if (j == 0) out[b] = red[0] + mb2[0];
}

// ---------------- launch ----------------
extern "C" void kernel_launch(void* const* d_in, const int* in_sizes, int n_in,
                              void* d_out, int out_size, void* d_ws, size_t ws_size,
                              hipStream_t stream) {
  (void)n_in; (void)ws_size;
  const float* x      = (const float*)d_in[0];
  const int*   eidx   = (const int*)  d_in[1];
  const int*   batch  = (const int*)  d_in[2];
  const float* W1     = (const float*)d_in[3];
  const float* b1     = (const float*)d_in[4];
  const float* W2     = (const float*)d_in[5];
  const float* b2     = (const float*)d_in[6];
  const float* gamma1 = (const float*)d_in[7];
  const float* beta1  = (const float*)d_in[8];
  const float* gamma2 = (const float*)d_in[9];
  const float* beta2  = (const float*)d_in[10];
  const float* aW1    = (const float*)d_in[11];
  const float* ab1    = (const float*)d_in[12];
  const float* aw2    = (const float*)d_in[13];
  const float* ab2    = (const float*)d_in[14];
  const float* mW1    = (const float*)d_in[15];
  const float* mb1    = (const float*)d_in[16];
  const float* mw2    = (const float*)d_in[17];
  const float* mb2    = (const float*)d_in[18];
  float* out = (float*)d_out;

  const int N  = in_sizes[0] / HD;
  const int E  = in_sizes[1] / 2;
  const int B  = out_size;
  const int NH = N * HD;
  const int* src = eidx;
  const int* dst = eidx + E;

  char* ws = (char*)d_ws;
  size_t off = 0;
  auto take = [&](size_t bytes) -> char* {
    char* p = ws + off;
    off += (bytes + 255) & ~(size_t)255;
    return p;
  };
  unsigned short* w1t  = (unsigned short*)take((size_t)HD * HD * 2);
  unsigned short* w2t  = (unsigned short*)take((size_t)HD * HD * 2);
  unsigned short* aw1t = (unsigned short*)take((size_t)HD * HD * 2);
  float* deg    = (float*)take((size_t)N * 4);
  float* dinv   = (float*)take((size_t)N * 4);
  float* bnsum  = (float*)take((size_t)2 * HD * 4);
  float* bnsumsq = bnsum + HD;
  float* score  = (float*)take((size_t)N * 4);
  float* eexp   = (float*)take((size_t)N * 4);
  float* segm   = (float*)take((size_t)B * 4);
  float* segd   = (float*)take((size_t)B * 4);
  float* g      = (float*)take((size_t)B * HD * 4);
  float* P0     = (float*)take((size_t)NH * 4);             // X@W / h@W / attention pre-tanh (f32)
  float* P1     = (float*)take((size_t)NH * 4);             // agg -> y (f32)
  float* P2     = (float*)take((size_t)NH * 4);             // activations h (f32, for pooling)
  unsigned short* Ab = (unsigned short*)take((size_t)NH * 2); // bf16 activations fed to WMMA GEMMs

  const int g256N  = (N + 255) / 256;
  const int g256E  = (E + 255) / 256;
  const int gWaveN = (N + 7) / 8;
  const int gGemm  = (N + 63) / 64;
  const int gNH4   = (NH / 4 + 255) / 256;

  // init
  fill_f32<<<g256N, 256, 0, stream>>>(deg, 0.f, N);
  fill_f32<<<(NH + 255) / 256, 256, 0, stream>>>(P1, 0.f, NH);
  fill_f32<<<1, 256, 0, stream>>>(bnsum, 0.f, 2 * HD);
  fill_f32<<<(B + 255) / 256, 256, 0, stream>>>(segm, -3.0e38f, B);
  fill_f32<<<(B + 255) / 256, 256, 0, stream>>>(segd, 0.f, B);
  fill_f32<<<(B * HD + 255) / 256, 256, 0, stream>>>(g, 0.f, B * HD);

  prep_w<<<64, 256, 0, stream>>>(W1,  w1t);
  prep_w<<<64, 256, 0, stream>>>(W2,  w2t);
  prep_w<<<64, 256, 0, stream>>>(aW1, aw1t);
  to_bf16<<<gNH4, 256, 0, stream>>>(x, Ab, NH / 4);

  degree_count<<<g256E, 256, 0, stream>>>(dst, deg, E);
  make_dinv<<<g256N, 256, 0, stream>>>(deg, dinv, N);

  // ---- GCN layer 1 ----
  gemm_bf16_wmma<<<gGemm, 256, 0, stream>>>(Ab, w1t, nullptr, P0, N, 0);
  edge_agg<<<(E + 7) / 8, 256, 0, stream>>>(src, dst, dinv, P0, P1, E);
  finish_stats<<<g256N, 128, 0, stream>>>(P0, P1, dinv, b1, bnsum, bnsumsq, N, 256);
  bn_apply_relu<<<(N * 32 + 255) / 256, 256, 0, stream>>>(P1, bnsum, bnsumsq, gamma1, beta1, P2, Ab, N);

  // ---- GCN layer 2 ----
  fill_f32<<<(NH + 255) / 256, 256, 0, stream>>>(P1, 0.f, NH);
  fill_f32<<<1, 256, 0, stream>>>(bnsum, 0.f, 2 * HD);
  gemm_bf16_wmma<<<gGemm, 256, 0, stream>>>(Ab, w2t, nullptr, P0, N, 0);
  edge_agg<<<(E + 7) / 8, 256, 0, stream>>>(src, dst, dinv, P0, P1, E);
  finish_stats<<<g256N, 128, 0, stream>>>(P0, P1, dinv, b2, bnsum, bnsumsq, N, 256);
  bn_apply_relu<<<(N * 32 + 255) / 256, 256, 0, stream>>>(P1, bnsum, bnsumsq, gamma2, beta2, P2, Ab, N);

  // ---- attention pooling ----
  gemm_bf16_wmma<<<gGemm, 256, 0, stream>>>(Ab, aw1t, ab1, P0, N, 1);
  score_kernel<<<gWaveN, 256, 0, stream>>>(P0, aw2, ab2, score, N);
  seg_max<<<g256N, 256, 0, stream>>>(score, batch, segm, N);
  seg_expsum<<<g256N, 256, 0, stream>>>(score, batch, segm, eexp, segd, N);
  pool_kernel<<<gWaveN, 256, 0, stream>>>(P2, eexp, segd, batch, g, N);

  // ---- scoring MLP ----
  final_mlp<<<B, 128, 0, stream>>>(g, mW1, mb1, mw2, mb2, out);
}
